// MPNN_Surrogate_88562225643709
// MI455X (gfx1250) — compile-verified, matrix operand order checked by reading, not run on hardware
//
#include <hip/hip_runtime.h>
#include <stdint.h>

// ---------------------------------------------------------------------------
// MPNN surrogate for MI455X (gfx1250): bf16 WMMA path, fused edge/node MLPs.
// Round 2: software-pipelined fragment fetches (B ring-4, A ring-2).
// ---------------------------------------------------------------------------

typedef __attribute__((ext_vector_type(16))) __bf16 v16bf;
typedef __attribute__((ext_vector_type(8)))  float  v8f;

#define HDIM 128
#define TS   136   // padded LDS row stride (ushorts) to dodge bank conflicts

static __device__ __forceinline__ unsigned short f2bf(float f) {
    unsigned int u = __float_as_uint(f);
    u += 0x7FFFu + ((u >> 16) & 1u);      // round-to-nearest-even
    return (unsigned short)(u >> 16);
}

union Frag {
    v16bf v;
    uint4 q[2];
    unsigned short s[16];
};

static __device__ __forceinline__ void loadFrag(Frag& f, const unsigned short* p) {
    f.q[0] = *(const uint4*)(p);
    f.q[1] = *(const uint4*)(p + 8);
}
static __device__ __forceinline__ void loadFragSplit(Frag& f, const unsigned short* p) {
    f.q[0] = *(const uint4*)(p);
    f.q[1] = *(const uint4*)(p + 16);
}

// ---------------------------------------------------------------------------
// Pack a row-major weight [Ktot, 128] (f32) into bf16 WMMA B-fragments.
// Fragment tile (kt, nt): 32 lanes x 16 bf16 contiguous per lane.
// Lane layout (ISA 7.12.2): n = nt*16 + (lane&15); kh = lane>>4;
//   elems 0..7  -> K = kt*32 + kh*8 + j
//   elems 8..15 -> K = kt*32 + 16 + kh*8 + (j-8)
// ---------------------------------------------------------------------------
__global__ void pack_weight_kernel(const float* __restrict__ W,
                                   unsigned short* __restrict__ dst, int total) {
    int tid = blockIdx.x * blockDim.x + threadIdx.x;
    if (tid >= total) return;
    int j    = tid & 15;
    int lane = (tid >> 4) & 31;
    int t    = tid >> 9;
    int nt   = t & 7;
    int kt   = t >> 3;
    int n    = nt * 16 + (lane & 15);
    int kh   = lane >> 4;
    int k    = kt * 32 + (j < 8 ? kh * 8 + j : 16 + kh * 8 + (j - 8));
    dst[tid] = f2bf(W[k * HDIM + n]);
}

// h = x @ enc_w + enc_b ; also produce bf16 mirror
__global__ void encoder_kernel(const float* __restrict__ x,
                               const float* __restrict__ ew,
                               const float* __restrict__ eb,
                               float* __restrict__ h,
                               unsigned short* __restrict__ hbf, int total) {
    int idx = blockIdx.x * blockDim.x + threadIdx.x;
    if (idx >= total) return;
    int n = idx >> 7, j = idx & 127;
    const float* xr = x + n * 16;
    float s = eb[j];
#pragma unroll
    for (int k = 0; k < 16; k++) s += xr[k] * ew[k * HDIM + j];
    h[idx] = s;
    hbf[idx] = f2bf(s);
}

__global__ void zero_kernel(float* __restrict__ p, int total) {
    int idx = blockIdx.x * blockDim.x + threadIdx.x;
    if (idx < total) p[idx] = 0.0f;
}

// out = h @ dec_w[128,3] + dec_b
__global__ void decoder_kernel(const float* __restrict__ h,
                               const float* __restrict__ dw,
                               const float* __restrict__ db,
                               float* __restrict__ out, int N) {
    int n = blockIdx.x * blockDim.x + threadIdx.x;
    if (n >= N) return;
    float s0 = db[0], s1 = db[1], s2 = db[2];
    const float* hr = h + n * HDIM;
#pragma unroll 4
    for (int k = 0; k < HDIM; k++) {
        float hv = hr[k];
        s0 += hv * dw[k * 3 + 0];
        s1 += hv * dw[k * 3 + 1];
        s2 += hv * dw[k * 3 + 2];
    }
    out[n * 3 + 0] = s0; out[n * 3 + 1] = s1; out[n * 3 + 2] = s2;
}

// ---------------------------------------------------------------------------
// Edge MLP, fused:  e = relu([h[row]||h[col]] @ W1 + b1) @ W2 + b2
// then agg[col] += e (f32 atomics). One wave handles 16 edges.
// ---------------------------------------------------------------------------
__global__ __launch_bounds__(256) void edge_kernel(
    const unsigned short* __restrict__ hbf, const long long* __restrict__ ei,
    const unsigned short* __restrict__ w1p, const float* __restrict__ b1,
    const unsigned short* __restrict__ w2p, const float* __restrict__ b2,
    float* __restrict__ agg, int E) {
    __shared__ unsigned short tile[8][16 * TS];
    const int lane = threadIdx.x & 31;
    const int wid  = threadIdx.x >> 5;
    const int base = (blockIdx.x * 8 + wid) * 16;
    if (base >= E) return;
    const int m0 = lane & 15;
    const int kh = lane >> 4;
    const long long src = ei[base + m0];
    const long long dst = ei[E + base + m0];
    const unsigned short* sf = hbf + src * HDIM;
    const unsigned short* df = hbf + dst * HDIM;

    v8f acc[8];
#pragma unroll
    for (int nt = 0; nt < 8; nt++) {
        float bv = b1[nt * 16 + m0];
#pragma unroll
        for (int i = 0; i < 8; i++) acc[nt][i] = bv;
    }
    // ---------------- GEMM1: K = 256 (src feats || dst feats) --------------
    {
        Frag a[2], b[4];
        loadFragSplit(a[0], sf + kh * 8);
        loadFrag(b[0], w1p + lane * 16);
        loadFrag(b[1], w1p + 512 + lane * 16);
#pragma unroll
        for (int kk = 0; kk < 8; kk++) {
            if (kk < 7) {
                const int kn = kk + 1;
                loadFragSplit(a[(kk + 1) & 1],
                              (kn < 4 ? sf : df) + (kn & 3) * 32 + kh * 8);
            }
#pragma unroll
            for (int nt = 0; nt < 8; nt++) {
                const int t = kk * 8 + nt;
                if (t + 2 < 64)
                    loadFrag(b[(t + 2) & 3], w1p + (t + 2) * 512 + lane * 16);
                acc[nt] = __builtin_amdgcn_wmma_f32_16x16x32_bf16(
                    false, a[kk & 1].v, false, b[t & 3].v, (short)0, acc[nt],
                    false, false);
            }
        }
    }
    // ReLU + repack C-layout -> A-layout through LDS (row-major 16 x 128)
    unsigned short* T = tile[wid];
#pragma unroll
    for (int nt = 0; nt < 8; nt++)
#pragma unroll
        for (int v = 0; v < 8; v++) {
            float xv = acc[nt][v];
            T[(v + kh * 8) * TS + nt * 16 + m0] = f2bf(xv > 0.f ? xv : 0.f);
        }
    v8f acc2[8];
#pragma unroll
    for (int nt = 0; nt < 8; nt++) {
        float bv = b2[nt * 16 + m0];
#pragma unroll
        for (int i = 0; i < 8; i++) acc2[nt][i] = bv;
    }
    // ---------------- GEMM2: K = 128 ---------------------------------------
    {
        Frag a[2], b[4];
        loadFragSplit(a[0], T + m0 * TS + kh * 8);
        loadFrag(b[0], w2p + lane * 16);
        loadFrag(b[1], w2p + 512 + lane * 16);
#pragma unroll
        for (int kk = 0; kk < 4; kk++) {
            if (kk < 3)
                loadFragSplit(a[(kk + 1) & 1],
                              T + m0 * TS + (kk + 1) * 32 + kh * 8);
#pragma unroll
            for (int nt = 0; nt < 8; nt++) {
                const int t = kk * 8 + nt;
                if (t + 2 < 32)
                    loadFrag(b[(t + 2) & 3], w2p + (t + 2) * 512 + lane * 16);
                acc2[nt] = __builtin_amdgcn_wmma_f32_16x16x32_bf16(
                    false, a[kk & 1].v, false, b[t & 3].v, (short)0, acc2[nt],
                    false, false);
            }
        }
    }
    // Fused scatter-add into agg[col]
#pragma unroll
    for (int v = 0; v < 8; v++) {
        const int m = v + kh * 8;
        const long long d = ei[E + base + m];
        float* ar = agg + d * HDIM + m0;
#pragma unroll
        for (int nt = 0; nt < 8; nt++)
            unsafeAtomicAdd(ar + nt * 16, acc2[nt][v]);
    }
}

// ---------------------------------------------------------------------------
// Node MLP, fused:  h += relu([h||agg] @ W1 + b1) @ W2 + b2  (+ bf16 mirror)
// One wave handles 16 consecutive nodes.
// ---------------------------------------------------------------------------
__global__ __launch_bounds__(256) void node_kernel(
    float* __restrict__ h, unsigned short* __restrict__ hbf,
    const float* __restrict__ agg,
    const unsigned short* __restrict__ w1p, const float* __restrict__ b1,
    const unsigned short* __restrict__ w2p, const float* __restrict__ b2,
    int N) {
    __shared__ unsigned short tile[8][16 * TS];
    const int lane = threadIdx.x & 31;
    const int wid  = threadIdx.x >> 5;
    const int base = (blockIdx.x * 8 + wid) * 16;
    if (base >= N) return;
    const int m0 = lane & 15;
    const int kh = lane >> 4;
    const int node0 = base + m0;
    const unsigned short* sf = hbf + node0 * HDIM;
    const float* af = agg + node0 * HDIM;

    v8f acc[8];
#pragma unroll
    for (int nt = 0; nt < 8; nt++) {
        float bv = b1[nt * 16 + m0];
#pragma unroll
        for (int i = 0; i < 8; i++) acc[nt][i] = bv;
    }
    // ---------------- GEMM1: K = 256 (h || agg) ----------------------------
    {
        Frag a[2], b[4];
        loadFragSplit(a[0], sf + kh * 8);
        loadFrag(b[0], w1p + lane * 16);
        loadFrag(b[1], w1p + 512 + lane * 16);
#pragma unroll
        for (int kk = 0; kk < 8; kk++) {
            if (kk < 7) {
                const int kn = kk + 1;
                Frag& an = a[(kk + 1) & 1];
                if (kn < 4) {
                    loadFragSplit(an, sf + kn * 32 + kh * 8);
                } else {
                    const float* g = af + (kn - 4) * 32 + kh * 8;
#pragma unroll
                    for (int j = 0; j < 8; j++) an.s[j] = f2bf(g[j]);
#pragma unroll
                    for (int j = 0; j < 8; j++) an.s[8 + j] = f2bf(g[16 + j]);
                }
            }
#pragma unroll
            for (int nt = 0; nt < 8; nt++) {
                const int t = kk * 8 + nt;
                if (t + 2 < 64)
                    loadFrag(b[(t + 2) & 3], w1p + (t + 2) * 512 + lane * 16);
                acc[nt] = __builtin_amdgcn_wmma_f32_16x16x32_bf16(
                    false, a[kk & 1].v, false, b[t & 3].v, (short)0, acc[nt],
                    false, false);
            }
        }
    }
    unsigned short* T = tile[wid];
#pragma unroll
    for (int nt = 0; nt < 8; nt++)
#pragma unroll
        for (int v = 0; v < 8; v++) {
            float xv = acc[nt][v];
            T[(v + kh * 8) * TS + nt * 16 + m0] = f2bf(xv > 0.f ? xv : 0.f);
        }
    v8f acc2[8];
#pragma unroll
    for (int nt = 0; nt < 8; nt++) {
        float bv = b2[nt * 16 + m0];
#pragma unroll
        for (int i = 0; i < 8; i++) acc2[nt][i] = bv;
    }
    // ---------------- GEMM2: K = 128 ---------------------------------------
    {
        Frag a[2], b[4];
        loadFragSplit(a[0], T + m0 * TS + kh * 8);
        loadFrag(b[0], w2p + lane * 16);
        loadFrag(b[1], w2p + 512 + lane * 16);
#pragma unroll
        for (int kk = 0; kk < 4; kk++) {
            if (kk < 3)
                loadFragSplit(a[(kk + 1) & 1],
                              T + m0 * TS + (kk + 1) * 32 + kh * 8);
#pragma unroll
            for (int nt = 0; nt < 8; nt++) {
                const int t = kk * 8 + nt;
                if (t + 2 < 32)
                    loadFrag(b[(t + 2) & 3], w2p + (t + 2) * 512 + lane * 16);
                acc2[nt] = __builtin_amdgcn_wmma_f32_16x16x32_bf16(
                    false, a[kk & 1].v, false, b[t & 3].v, (short)0, acc2[nt],
                    false, false);
            }
        }
    }
    // residual writeback + bf16 mirror
#pragma unroll
    for (int v = 0; v < 8; v++) {
        const int m = v + kh * 8;
        const int node = base + m;
        float* hr = h + node * HDIM;
#pragma unroll
        for (int nt = 0; nt < 8; nt++) {
            int n = nt * 16 + m0;
            float nv = hr[n] + acc2[nt][v];
            hr[n] = nv;
            hbf[node * HDIM + n] = f2bf(nv);
        }
    }
}

// ---------------------------------------------------------------------------
extern "C" void kernel_launch(void* const* d_in, const int* in_sizes, int n_in,
                              void* d_out, int out_size, void* d_ws, size_t ws_size,
                              hipStream_t stream) {
    const float*     x       = (const float*)d_in[0];
    const long long* ei      = (const long long*)d_in[1];
    const float*     enc_w   = (const float*)d_in[2];
    const float*     enc_b   = (const float*)d_in[3];
    const float*     dec_w   = (const float*)d_in[4];
    const float*     dec_b   = (const float*)d_in[5];
    const float*     edge_w1 = (const float*)d_in[6];
    const float*     edge_b1 = (const float*)d_in[7];
    const float*     edge_w2 = (const float*)d_in[8];
    const float*     edge_b2 = (const float*)d_in[9];
    const float*     node_w1 = (const float*)d_in[10];
    const float*     node_b1 = (const float*)d_in[11];
    const float*     node_w2 = (const float*)d_in[12];
    const float*     node_b2 = (const float*)d_in[13];

    const int N = in_sizes[0] / 16;   // 50000
    const int E = in_sizes[1] / 2;    // 800000
    const int NH = N * HDIM;

    // workspace carve-up (256B aligned slices)
    size_t off = 0;
    auto carve = [&](size_t bytes) -> void* {
        void* p = (char*)d_ws + off;
        off = (off + bytes + 255) & ~(size_t)255;
        return p;
    };
    float*          h    = (float*)carve((size_t)NH * 4);
    unsigned short* hbf  = (unsigned short*)carve((size_t)NH * 2);
    float*          agg  = (float*)carve((size_t)NH * 4);
    unsigned short* ew1p = (unsigned short*)carve((size_t)3 * 256 * HDIM * 2);
    unsigned short* ew2p = (unsigned short*)carve((size_t)3 * HDIM * HDIM * 2);
    unsigned short* nw1p = (unsigned short*)carve((size_t)3 * 256 * HDIM * 2);
    unsigned short* nw2p = (unsigned short*)carve((size_t)3 * HDIM * HDIM * 2);

    // pack all weights into WMMA B-fragment layout (bf16)
    {
        int t1 = 3 * 256 * HDIM, t2 = 3 * HDIM * HDIM;
        pack_weight_kernel<<<(t1 + 255) / 256, 256, 0, stream>>>(edge_w1, ew1p, t1);
        pack_weight_kernel<<<(t2 + 255) / 256, 256, 0, stream>>>(edge_w2, ew2p, t2);
        pack_weight_kernel<<<(t1 + 255) / 256, 256, 0, stream>>>(node_w1, nw1p, t1);
        pack_weight_kernel<<<(t2 + 255) / 256, 256, 0, stream>>>(node_w2, nw2p, t2);
    }

    encoder_kernel<<<(NH + 255) / 256, 256, 0, stream>>>(x, enc_w, enc_b, h, hbf, NH);

    const int edgeBlocks = (E / 16 + 7) / 8;
    const int nodeBlocks = (N / 16 + 7) / 8;
    for (int l = 0; l < 3; l++) {
        zero_kernel<<<(NH + 255) / 256, 256, 0, stream>>>(agg, NH);
        edge_kernel<<<edgeBlocks, 256, 0, stream>>>(
            hbf, ei,
            ew1p + (size_t)l * 256 * HDIM, edge_b1 + l * HDIM,
            ew2p + (size_t)l * HDIM * HDIM, edge_b2 + l * HDIM,
            agg, E);
        node_kernel<<<nodeBlocks, 256, 0, stream>>>(
            h, hbf, agg,
            nw1p + (size_t)l * 256 * HDIM, node_b1 + l * HDIM,
            nw2p + (size_t)l * HDIM * HDIM, node_b2 + l * HDIM,
            N);
    }

    decoder_kernel<<<(N + 255) / 256, 256, 0, stream>>>(h, dec_w, dec_b, (float*)d_out, N);
}
